// BiLSTMCRFCharCNN_39694087749666
// MI455X (gfx1250) — matrix operand order repaired and checked by reading, not compiled, not run on hardware
//
#include <hip/hip_runtime.h>
#include <hip/hip_bf16.h>

// ---------------------------------------------------------------------------
// BiLSTM-CRF + CharCNN for MI455X (gfx1250, wave32, WMMA bf16).
//
// Pipeline:
//   1. cvt weights f32->bf16 (padded), fuse LSTM biases
//   2. embed + char-CNN  -> combined [S*B][480] bf16  (rows r = s*32 + b)
//   3. WMMA GEMM (32x64 tile per wave): Gpre[dir] = combined @ wih^T + bias
//   4. lstm_scan: 1 workgroup (32 waves) per direction; h in LDS, c in regs,
//      per-step 32x2048x512 GEMM via v_wmma_f32_16x16x32_bf16
//   5. repeat 3+4 for layer 1; highway GEMMs (sigmoid/relu epilogue) + elementwise
//   6. FC GEMM (N padded 20->64) -> logits f32 [8192][64]
//   7. CRF forward scan + gold score + aux NLL -> scalar loss
// ---------------------------------------------------------------------------

#define DEVI __device__ __forceinline__

typedef __attribute__((ext_vector_type(16))) __bf16 bf16x16;
typedef __attribute__((ext_vector_type(8)))  __bf16 bf16x8;
typedef __attribute__((ext_vector_type(8)))  float  f32x8;

DEVI f32x8 zero8() {
  f32x8 z;
#pragma unroll
  for (int i = 0; i < 8; ++i) z[i] = 0.0f;
  return z;
}

DEVI float sigmf(float x) { return 1.0f / (1.0f + __expf(-x)); }

DEVI f32x8 wmma_bf16(bf16x16 a, bf16x16 b, f32x8 c) {
  return __builtin_amdgcn_wmma_f32_16x16x32_bf16(
      /*neg_a=*/false, a, /*neg_b=*/false, b,
      /*c_mod=*/(short)0, c, /*reuse_a=*/false, /*reuse_b=*/false);
}

// A fragment: 16x32 bf16 (M x K). ISA layout: lanes 0-15 hold K[kc..kc+7] then
// K[kc+16..kc+23]; lanes 16-31 hold K[kc+8..kc+15] then K[kc+24..kc+31].
DEVI bf16x16 frag_a(const __bf16* base, int ld, int row0, int kc, int lane) {
  const __bf16* p = base + (size_t)(row0 + (lane & 15)) * ld + kc + ((lane >> 4) & 1) * 8;
  bf16x8 lo = *(const bf16x8*)(p);
  bf16x8 hi = *(const bf16x8*)(p + 16);
  return __builtin_shufflevector(lo, hi, 0,1,2,3,4,5,6,7,8,9,10,11,12,13,14,15);
}

// B fragment: 32x16 bf16 (K x N) from row-major W[N][K] (B = W^T). ISA layout:
// lanes 0-15 = col, K[kc..kc+7] then K[kc+16..kc+23]; lanes 16-31 = col, K[kc+8..kc+15] then K[kc+24..kc+31].
DEVI bf16x16 frag_b(const __bf16* W, int ld, int col0, int kc, int lane) {
  const __bf16* p = W + (size_t)(col0 + (lane & 15)) * ld + kc + ((lane >> 4) & 1) * 8;
  bf16x8 lo = *(const bf16x8*)(p);
  bf16x8 hi = *(const bf16x8*)(p + 16);
  return __builtin_shufflevector(lo, hi, 0,1,2,3,4,5,6,7,8,9,10,11,12,13,14,15);
}

// --------------------------- small utility kernels -------------------------

__global__ void cvt_bf16_pad(const float* __restrict__ src, __bf16* __restrict__ dst,
                             int dstRows, int srcRows, int dstCols, int srcCols) {
  int total = dstRows * dstCols;
  for (int i = blockIdx.x * blockDim.x + threadIdx.x; i < total;
       i += gridDim.x * blockDim.x) {
    int r = i / dstCols, c = i - r * dstCols;
    float v = (r < srcRows && c < srcCols) ? src[(size_t)r * srcCols + c] : 0.0f;
    dst[i] = (__bf16)v;
  }
}

__global__ void add_vec(const float* __restrict__ a, const float* __restrict__ b,
                        float* __restrict__ o, int n) {
  for (int i = blockIdx.x * blockDim.x + threadIdx.x; i < n; i += gridDim.x * blockDim.x)
    o[i] = a[i] + b[i];
}

__global__ void pad_vec(const float* __restrict__ a, float* __restrict__ o, int n, int npad) {
  for (int i = blockIdx.x * blockDim.x + threadIdx.x; i < npad; i += gridDim.x * blockDim.x)
    o[i] = (i < n) ? a[i] : 0.0f;
}

// --------------------------- embedding + char CNN --------------------------
// One block per word; combined row r = s*32 + b, layout [S*B][480] bf16:
// [0..299] word embed, [300..449] char-CNN (k=2,3,4 x 50), [450..479] zero pad.

__global__ __launch_bounds__(160)
void embed_charcnn(const int* __restrict__ x, const int* __restrict__ xc,
                   const float* __restrict__ wemb, const float* __restrict__ cet,
                   const float* __restrict__ w2, const float* __restrict__ b2,
                   const float* __restrict__ w3, const float* __restrict__ b3,
                   const float* __restrict__ w4, const float* __restrict__ b4,
                   __bf16* __restrict__ comb) {
  __shared__ float ce[16][30];
  int r = blockIdx.x;          // r = s*32 + b
  int s = r >> 5, b = r & 31;
  int tid = threadIdx.x;
  for (int i = tid; i < 16 * 30; i += blockDim.x) {
    int ch = i / 30, c = i - ch * 30;
    int ci = xc[((size_t)b * 256 + s) * 16 + ch];
    ce[ch][c] = cet[(size_t)ci * 30 + c];
  }
  __syncthreads();
  __bf16* out = comb + (size_t)r * 480;
  int wid = x[b * 256 + s];
  for (int i = tid; i < 300; i += blockDim.x)
    out[i] = (__bf16)wemb[(size_t)wid * 300 + i];
  if (tid < 30) out[450 + tid] = (__bf16)0.0f;
  if (tid < 150) {
    int kidx = tid / 50, f = tid - kidx * 50;
    int kk = kidx + 2;
    const float* wp = (kidx == 0 ? w2 : (kidx == 1 ? w3 : w4)) + (size_t)f * 30 * kk;
    float bias = (kidx == 0 ? b2 : (kidx == 1 ? b3 : b4))[f];
    float mx = -1e30f;
    for (int p = 0; p <= 16 - kk; ++p) {
      float acc = 0.0f;
      for (int c = 0; c < 30; ++c)
        for (int j = 0; j < kk; ++j)
          acc += ce[p + j][c] * wp[c * kk + j];
      mx = fmaxf(mx, acc + bias);
    }
    out[300 + kidx * 50 + f] = (__bf16)fmaxf(mx, 0.0f);
  }
}

// ------------------------------- WMMA GEMM ---------------------------------
// C[M][N] = act( A[M][K]bf16 @ W[N][K]bf16^T + bias[N] ); out f32 (Cf) or bf16 (Cb).
// One wave computes a 32x64 C tile (2 M-tiles x 4 N-tiles): each A fragment is
// reused 4x, each B fragment 2x -> 1.5 b128 loads per WMMA. N must be a
// multiple of 64, M a multiple of 32. act: 0 none, 1 relu, 2 sigmoid.

__global__ __launch_bounds__(256)
void wmma_gemm(const __bf16* __restrict__ A, int lda,
               const __bf16* __restrict__ W, int ldw,
               const float* __restrict__ bias,
               float* __restrict__ Cf, __bf16* __restrict__ Cb, int ldc,
               int M, int N, int K, int act) {
  int lane = threadIdx.x & 31;
  int wave = threadIdx.x >> 5;
  int gw = blockIdx.x * (blockDim.x >> 5) + wave;
  int ntiles = N >> 6;                       // 64-wide wave tiles
  int mb = gw / ntiles, nb = gw - mb * ntiles;
  if (mb >= (M >> 5)) return;                // wave-uniform

  f32x8 acc[4][2];
#pragma unroll
  for (int nt = 0; nt < 4; ++nt) { acc[nt][0] = zero8(); acc[nt][1] = zero8(); }

  for (int kc = 0; kc < K; kc += 32) {
    bf16x16 a0 = frag_a(A, lda, mb * 32, kc, lane);
    bf16x16 a1 = frag_a(A, lda, mb * 32 + 16, kc, lane);
#pragma unroll
    for (int nt = 0; nt < 4; ++nt) {
      bf16x16 bf = frag_b(W, ldw, nb * 64 + nt * 16, kc, lane);
      acc[nt][0] = wmma_bf16(a0, bf, acc[nt][0]);
      acc[nt][1] = wmma_bf16(a1, bf, acc[nt][1]);
    }
  }

  int hi = (lane >> 4) & 1;
#pragma unroll
  for (int nt = 0; nt < 4; ++nt) {
    int n = nb * 64 + nt * 16 + (lane & 15);
    float bv = bias ? bias[n] : 0.0f;
#pragma unroll
    for (int e = 0; e < 8; ++e) {
      int m0 = mb * 32 + 8 * hi + e;
      float v0 = acc[nt][0][e] + bv;
      float v1 = acc[nt][1][e] + bv;
      if (act == 1) { v0 = fmaxf(v0, 0.0f); v1 = fmaxf(v1, 0.0f); }
      else if (act == 2) { v0 = sigmf(v0); v1 = sigmf(v1); }
      if (Cf) {
        Cf[(size_t)m0 * ldc + n] = v0;
        Cf[(size_t)(m0 + 16) * ldc + n] = v1;
      } else {
        Cb[(size_t)m0 * ldc + n] = (__bf16)v0;
        Cb[(size_t)(m0 + 16) * ldc + n] = (__bf16)v1;
      }
    }
  }
}

// ------------------------------- LSTM scan ---------------------------------
// grid.x = 2 (direction). 32 waves; wave w owns hidden cols [16w,16w+16).
// Per step: gates(32x2048) = Gpre[s] + h @ whh^T via WMMA; h in LDS (bf16),
// c in registers. H layout: [S*32][1024], dir0 -> cols 0..511, dir1 -> 512..1023.

__global__ __launch_bounds__(1024)
void lstm_scan(const __bf16* __restrict__ Gpre,  // [2][S*32][2048] bf16
               const __bf16* __restrict__ whh,   // [2][2048][512] bf16
               __bf16* __restrict__ H,           // [S*32][1024] bf16
               int S) {
  __shared__ __align__(16) __bf16 hbuf[32 * 512];
  int dir = blockIdx.x;
  int lane = threadIdx.x & 31;
  int wave = threadIdx.x >> 5;
  for (int i = threadIdx.x; i < 32 * 512; i += blockDim.x) hbuf[i] = (__bf16)0.0f;
  __syncthreads();

  int hi = (lane >> 4) & 1;
  int col = (wave << 4) + (lane & 15);
  const __bf16* Wd = whh + (size_t)dir * 2048 * 512;
  const __bf16* Gd = Gpre + (size_t)dir * S * 32 * 2048;

  float c0[8], c1[8];
#pragma unroll
  for (int e = 0; e < 8; ++e) { c0[e] = 0.0f; c1[e] = 0.0f; }

  for (int t = 0; t < S; ++t) {
    int s = dir ? (S - 1 - t) : t;
    f32x8 acc[4][2];
#pragma unroll
    for (int g = 0; g < 4; ++g) { acc[g][0] = zero8(); acc[g][1] = zero8(); }

    for (int kc = 0; kc < 512; kc += 32) {
      bf16x16 a0 = frag_a(hbuf, 512, 0, kc, lane);
      bf16x16 a1 = frag_a(hbuf, 512, 16, kc, lane);
#pragma unroll
      for (int g = 0; g < 4; ++g) {
        bf16x16 bf = frag_b(Wd, 512, g * 512 + (wave << 4), kc, lane);
        acc[g][0] = wmma_bf16(a0, bf, acc[g][0]);
        acc[g][1] = wmma_bf16(a1, bf, acc[g][1]);
      }
    }

    const __bf16* grow = Gd + (size_t)s * 32 * 2048;
    float hn0[8], hn1[8];
#pragma unroll
    for (int e = 0; e < 8; ++e) {
      int b0 = 8 * hi + e;
      int b1 = b0 + 16;
      float i0 = acc[0][0][e] + (float)grow[(size_t)b0 * 2048 + col];
      float f0 = acc[1][0][e] + (float)grow[(size_t)b0 * 2048 + 512 + col];
      float g0 = acc[2][0][e] + (float)grow[(size_t)b0 * 2048 + 1024 + col];
      float o0 = acc[3][0][e] + (float)grow[(size_t)b0 * 2048 + 1536 + col];
      float cn0 = sigmf(f0) * c0[e] + sigmf(i0) * tanhf(g0);
      c0[e] = cn0;
      hn0[e] = sigmf(o0) * tanhf(cn0);

      float i1 = acc[0][1][e] + (float)grow[(size_t)b1 * 2048 + col];
      float f1 = acc[1][1][e] + (float)grow[(size_t)b1 * 2048 + 512 + col];
      float g1 = acc[2][1][e] + (float)grow[(size_t)b1 * 2048 + 1024 + col];
      float o1 = acc[3][1][e] + (float)grow[(size_t)b1 * 2048 + 1536 + col];
      float cn1 = sigmf(f1) * c1[e] + sigmf(i1) * tanhf(g1);
      c1[e] = cn1;
      hn1[e] = sigmf(o1) * tanhf(cn1);
    }

    __syncthreads();   // all waves done reading hbuf for this step
#pragma unroll
    for (int e = 0; e < 8; ++e) {
      int b0 = 8 * hi + e;
      int b1 = b0 + 16;
      hbuf[b0 * 512 + col] = (__bf16)hn0[e];
      hbuf[b1 * 512 + col] = (__bf16)hn1[e];
      H[(size_t)(s * 32 + b0) * 1024 + dir * 512 + col] = (__bf16)hn0[e];
      H[(size_t)(s * 32 + b1) * 1024 + dir * 512 + col] = (__bf16)hn1[e];
    }
    __syncthreads();   // new h visible before next step's reads
  }
}

// ------------------------- highway elementwise -----------------------------

__global__ void highway_ew(const __bf16* __restrict__ H1, const __bf16* __restrict__ T,
                           const __bf16* __restrict__ Hh, __bf16* __restrict__ o, int n) {
  for (int i = blockIdx.x * blockDim.x + threadIdx.x; i < n; i += gridDim.x * blockDim.x) {
    float t = (float)T[i];            // sigmoid already applied
    float h = (float)Hh[i];           // relu already applied
    float v = (float)H1[i];
    o[i] = (__bf16)(t * h + (1.0f - t) * v);
  }
}

// ------------------------------ CRF + aux loss -----------------------------
// logits: [S*32][64] f32 (cols 0..19 valid). Single block of 1024 threads.

__global__ __launch_bounds__(1024)
void crf_kernel(const float* __restrict__ logits,
                const int* __restrict__ x, const int* __restrict__ tags,
                const float* __restrict__ cstart, const float* __restrict__ cend,
                const float* __restrict__ ctrans, float* __restrict__ out) {
  __shared__ float trans[400];
  __shared__ float alpha[32][20];
  __shared__ float redA[1024], redB[1024];
  __shared__ float scoreArr[32], logZ[32];
  int tid = threadIdx.x;
  for (int i = tid; i < 400; i += 1024) trans[i] = ctrans[i];
  __syncthreads();

  // aux NLL (log-softmax over 20 tags, masked mean)
  float nll = 0.0f, cnt = 0.0f;
  for (int r = tid; r < 8192; r += 1024) {
    int s = r >> 5, b = r & 31;
    int tg = tags[b * 256 + s];
    if (tg != 0) {
      const float* lr = logits + (size_t)r * 64;
      float m = lr[0];
      for (int t = 1; t < 20; ++t) m = fmaxf(m, lr[t]);
      float sum = 0.0f;
      for (int t = 0; t < 20; ++t) sum += __expf(lr[t] - m);
      nll += (m + __logf(sum)) - lr[tg];
      cnt += 1.0f;
    }
  }
  redA[tid] = nll; redB[tid] = cnt;
  __syncthreads();
  for (int off = 512; off > 0; off >>= 1) {
    if (tid < off) { redA[tid] += redA[tid + off]; redB[tid] += redB[tid + off]; }
    __syncthreads();
  }

  // gold path scores
  if (tid < 32) {
    int b = tid;
    int t0 = tags[b * 256];
    float sc = cstart[t0] + logits[(size_t)b * 64 + t0];  // s=0 row = b
    int prev = t0;
    int len = (x[b * 256] != 0) ? 1 : 0;
    for (int s = 1; s < 256; ++s) {
      int tg = tags[b * 256 + s];
      if (x[b * 256 + s] != 0) {
        sc += trans[prev * 20 + tg] + logits[(size_t)(s * 32 + b) * 64 + tg];
        ++len;
      }
      prev = tg;
    }
    int lastTag = tags[b * 256 + (len - 1)];
    scoreArr[b] = sc + cend[lastTag];
  }

  // forward algorithm
  int b = tid / 20, j = tid - b * 20;
  bool act = tid < 640;
  if (act) alpha[b][j] = cstart[j] + logits[(size_t)b * 64 + j];
  __syncthreads();
  for (int s = 1; s < 256; ++s) {
    float nv = 0.0f;
    if (act) {
      float em = logits[(size_t)(s * 32 + b) * 64 + j];
      float m = -1e30f;
      for (int i = 0; i < 20; ++i) m = fmaxf(m, alpha[b][i] + trans[i * 20 + j]);
      float sum = 0.0f;
      for (int i = 0; i < 20; ++i) sum += __expf(alpha[b][i] + trans[i * 20 + j] - m);
      float lse = m + __logf(sum) + em;
      nv = (x[b * 256 + s] != 0) ? lse : alpha[b][j];
    }
    __syncthreads();
    if (act) alpha[b][j] = nv;
    __syncthreads();
  }
  if (tid < 32) {
    float m = -1e30f;
    for (int t = 0; t < 20; ++t) m = fmaxf(m, alpha[tid][t] + cend[t]);
    float sum = 0.0f;
    for (int t = 0; t < 20; ++t) sum += __expf(alpha[tid][t] + cend[t] - m);
    logZ[tid] = m + __logf(sum);
  }
  __syncthreads();
  if (tid == 0) {
    float acc = 0.0f;
    for (int bb = 0; bb < 32; ++bb) acc += scoreArr[bb] - logZ[bb];
    float auxLoss = redA[0] / redB[0];
    out[0] = -(acc / 32.0f) + 0.1f * auxLoss;
  }
}

// ------------------------------- launcher ----------------------------------

extern "C" void kernel_launch(void* const* d_in, const int* in_sizes, int n_in,
                              void* d_out, int out_size, void* d_ws, size_t ws_size,
                              hipStream_t stream) {
  (void)in_sizes; (void)n_in; (void)out_size; (void)ws_size;
  const int*   x       = (const int*)d_in[0];
  const int*   xch     = (const int*)d_in[1];
  const int*   tags    = (const int*)d_in[2];
  const float* wemb    = (const float*)d_in[3];
  const float* cembT   = (const float*)d_in[4];
  const float* w2      = (const float*)d_in[5];
  const float* b2      = (const float*)d_in[6];
  const float* w3      = (const float*)d_in[7];
  const float* b3      = (const float*)d_in[8];
  const float* w4      = (const float*)d_in[9];
  const float* b4      = (const float*)d_in[10];
  const float* l0_wih  = (const float*)d_in[11];
  const float* l0_whh  = (const float*)d_in[12];
  const float* l0_bih  = (const float*)d_in[13];
  const float* l0_bhh  = (const float*)d_in[14];
  const float* l1_wih  = (const float*)d_in[15];
  const float* l1_whh  = (const float*)d_in[16];
  const float* l1_bih  = (const float*)d_in[17];
  const float* l1_bhh  = (const float*)d_in[18];
  const float* hw_Hw   = (const float*)d_in[19];
  const float* hw_Hb   = (const float*)d_in[20];
  const float* hw_Tw   = (const float*)d_in[21];
  const float* hw_Tb   = (const float*)d_in[22];
  const float* fc_w    = (const float*)d_in[23];
  const float* fc_b    = (const float*)d_in[24];
  const float* crf_s   = (const float*)d_in[25];
  const float* crf_e   = (const float*)d_in[26];
  const float* crf_t   = (const float*)d_in[27];

  char* ws = (char*)d_ws;
  size_t off = 0;
  auto alloc = [&](size_t bytes) -> char* {
    char* p = ws + off;
    off += (bytes + 255) & ~(size_t)255;
    return p;
  };

  __bf16* comb  = (__bf16*)alloc(8192ull * 480 * 2);
  __bf16* w0b   = (__bf16*)alloc(4096ull * 480 * 2);
  __bf16* wh0b  = (__bf16*)alloc(4096ull * 512 * 2);
  __bf16* w1b   = (__bf16*)alloc(4096ull * 1024 * 2);
  __bf16* wh1b  = (__bf16*)alloc(4096ull * 512 * 2);
  __bf16* hwTw  = (__bf16*)alloc(1024ull * 1024 * 2);
  __bf16* hwHw  = (__bf16*)alloc(1024ull * 1024 * 2);
  __bf16* fcwb  = (__bf16*)alloc(64ull * 1024 * 2);
  float*  bias0 = (float*)alloc(4096ull * 4);
  float*  bias1 = (float*)alloc(4096ull * 4);
  float*  fcbp  = (float*)alloc(64ull * 4);
  __bf16* Gpre  = (__bf16*)alloc(2ull * 8192 * 2048 * 2);
  __bf16* H0    = (__bf16*)alloc(8192ull * 1024 * 2);
  __bf16* H1    = (__bf16*)alloc(8192ull * 1024 * 2);
  __bf16* Tl    = (__bf16*)alloc(8192ull * 1024 * 2);
  __bf16* Hl    = (__bf16*)alloc(8192ull * 1024 * 2);
  __bf16* HWo   = (__bf16*)alloc(8192ull * 1024 * 2);
  float*  logit = (float*)alloc(8192ull * 64 * 4);

  // 1. weight conversion + bias fusion
  cvt_bf16_pad<<<1024, 256, 0, stream>>>(l0_wih, w0b, 4096, 4096, 480, 450);
  cvt_bf16_pad<<<1024, 256, 0, stream>>>(l0_whh, wh0b, 4096, 4096, 512, 512);
  cvt_bf16_pad<<<1024, 256, 0, stream>>>(l1_wih, w1b, 4096, 4096, 1024, 1024);
  cvt_bf16_pad<<<1024, 256, 0, stream>>>(l1_whh, wh1b, 4096, 4096, 512, 512);
  cvt_bf16_pad<<<1024, 256, 0, stream>>>(hw_Tw, hwTw, 1024, 1024, 1024, 1024);
  cvt_bf16_pad<<<1024, 256, 0, stream>>>(hw_Hw, hwHw, 1024, 1024, 1024, 1024);
  cvt_bf16_pad<<<64, 256, 0, stream>>>(fc_w, fcwb, 64, 20, 1024, 1024);
  add_vec<<<16, 256, 0, stream>>>(l0_bih, l0_bhh, bias0, 4096);
  add_vec<<<16, 256, 0, stream>>>(l1_bih, l1_bhh, bias1, 4096);
  pad_vec<<<1, 64, 0, stream>>>(fc_b, fcbp, 20, 64);

  // 2. embeddings + char CNN
  embed_charcnn<<<8192, 160, 0, stream>>>(x, xch, wemb, cembT, w2, b2, w3, b3, w4, b4, comb);

  auto gemm = [&](const __bf16* A, int lda, const __bf16* W, int ldw, const float* bias,
                  float* Cf, __bf16* Cb, int ldc, int M, int N, int K, int act) {
    int tiles = (M / 32) * (N / 64);
    wmma_gemm<<<dim3(tiles / 8), dim3(256), 0, stream>>>(A, lda, W, ldw, bias, Cf, Cb,
                                                         ldc, M, N, K, act);
  };

  // 3. layer 0
  gemm(comb, 480, w0b,               480, bias0,        nullptr, Gpre,                 2048, 8192, 2048, 480, 0);
  gemm(comb, 480, w0b + 2048 * 480,  480, bias0 + 2048, nullptr, Gpre + 8192ull * 2048, 2048, 8192, 2048, 480, 0);
  lstm_scan<<<2, 1024, 0, stream>>>(Gpre, wh0b, H0, 256);

  // 4. layer 1
  gemm(H0, 1024, w1b,                1024, bias1,        nullptr, Gpre,                 2048, 8192, 2048, 1024, 0);
  gemm(H0, 1024, w1b + 2048 * 1024,  1024, bias1 + 2048, nullptr, Gpre + 8192ull * 2048, 2048, 8192, 2048, 1024, 0);
  lstm_scan<<<2, 1024, 0, stream>>>(Gpre, wh1b, H1, 256);

  // 5. highway
  gemm(H1, 1024, hwTw, 1024, hw_Tb, nullptr, Tl, 1024, 8192, 1024, 1024, 2);  // sigmoid
  gemm(H1, 1024, hwHw, 1024, hw_Hb, nullptr, Hl, 1024, 8192, 1024, 1024, 1);  // relu
  highway_ew<<<2048, 256, 0, stream>>>(H1, Tl, Hl, HWo, 8192 * 1024);

  // 6. FC -> logits (N padded to 64)
  gemm(HWo, 1024, fcwb, 1024, fcbp, logit, nullptr, 64, 8192, 64, 1024, 0);

  // 7. CRF + aux loss -> scalar
  crf_kernel<<<1, 1024, 0, stream>>>(logit, x, tags, crf_s, crf_e, crf_t, (float*)d_out);
}
